// ResnetBlock_1597727834252
// MI455X (gfx1250) — matrix operand order repaired and checked by reading, not compiled, not run on hardware
//
#include <hip/hip_runtime.h>
#include <hip/hip_bf16.h>

// ------------------------- problem constants -------------------------
constexpr int   V    = 49152;
constexpr int   CIN  = 128;
constexpr int   COUT = 256;
constexpr int   KCH  = 8;
constexpr int   EDG  = V * 9;
constexpr float EPSG = 1e-5f;

typedef __bf16 bf16_t;
typedef __attribute__((ext_vector_type(16))) __bf16 v16bf;
typedef __attribute__((ext_vector_type(8)))  float  v8f;

__device__ __forceinline__ float silu_f(float x) {
  return x / (1.0f + __expf(-x));
}

// ------------------------- elementwise kernels -------------------------
__global__ void fill_zero_k(float* p, size_t n) {
  size_t stride = (size_t)gridDim.x * blockDim.x;
  for (size_t i = (size_t)blockIdx.x * blockDim.x + threadIdx.x; i < n; i += stride)
    p[i] = 0.0f;
}

// dst[i] = rowv[i & fmask]  (broadcast a length-(fmask+1) row over all vertices)
__global__ void bcast_row_k(float* __restrict__ dst, const float* __restrict__ rowv,
                            unsigned fmask, size_t n) {
  size_t stride = (size_t)gridDim.x * blockDim.x;
  for (size_t i = (size_t)blockIdx.x * blockDim.x + threadIdx.x; i < n; i += stride)
    dst[i] = rowv[(unsigned)i & fmask];
}

// ------------------------- CSR build (per call, on device) -------------------------
__global__ void count_rows_k(const int* __restrict__ row, int* __restrict__ counts, int n) {
  size_t stride = (size_t)gridDim.x * blockDim.x;
  for (size_t i = (size_t)blockIdx.x * blockDim.x + threadIdx.x; i < (size_t)n; i += stride)
    atomicAdd(&counts[row[i]], 1);
}

// exclusive scan of counts[V] -> rowptr[V+1]; single block of 1024, chunk = V/1024 = 48
__global__ __launch_bounds__(1024) void rowptr_scan_k(const int* __restrict__ counts,
                                                      int* __restrict__ rowptr) {
  __shared__ int sh[1024];
  const int t = threadIdx.x;
  const int chunk = V / 1024;
  const int base = t * chunk;
  int s = 0;
  for (int i = 0; i < chunk; ++i) s += counts[base + i];
  sh[t] = s;
  __syncthreads();
  for (int off = 1; off < 1024; off <<= 1) {   // Hillis-Steele inclusive scan
    int v = (t >= off) ? sh[t - off] : 0;
    __syncthreads();
    sh[t] += v;
    __syncthreads();
  }
  int run = (t == 0) ? 0 : sh[t - 1];          // exclusive prefix of this chunk
  for (int i = 0; i < chunk; ++i) {
    int c = counts[base + i];
    rowptr[base + i] = run;
    run += c;
  }
  if (t == 1023) rowptr[V] = run;              // == EDG
}

// cursor[] starts as a copy of rowptr[0..V-1]
__global__ void scatter_perm_k(const int* __restrict__ row, int* __restrict__ cursor,
                               int* __restrict__ perm, int n) {
  size_t stride = (size_t)gridDim.x * blockDim.x;
  for (size_t i = (size_t)blockIdx.x * blockDim.x + threadIdx.x; i < (size_t)n; i += stride) {
    int p = atomicAdd(&cursor[row[i]], 1);
    perm[p] = (int)i;
  }
}

// ------------------------- gather SpMV (no atomics) -------------------------
// out[v, c4..c4+3] = beta*initsrc[v, c4..] + alpha * sum_{j in row v} vals[e] * in[col[e], c4..]
// one thread per (vertex, 4-channel chunk); chunks = C/4 = 1<<cshift.
// out may alias initsrc (each thread reads its init before writing).
__global__ void spmv_csr_k(float* __restrict__ out, const float* __restrict__ in,
                           const float* initsrc, float beta, float alpha,
                           const int* __restrict__ rowptr, const int* __restrict__ perm,
                           const int* __restrict__ col, const float* __restrict__ vals,
                           int cshift) {
  const int chunks = 1 << cshift;
  size_t idx = (size_t)blockIdx.x * blockDim.x + threadIdx.x;
  if (idx >= (size_t)V * chunks) return;
  const int v  = (int)(idx >> cshift);
  const int c4 = ((int)idx & (chunks - 1)) << 2;
  const int C  = chunks << 2;
  const int jbeg = rowptr[v], jend = rowptr[v + 1];
  float4 acc = make_float4(0.f, 0.f, 0.f, 0.f);
  if (beta != 0.0f) {
    float4 iv = *(const float4*)(initsrc + (size_t)v * C + c4);
    acc.x = beta * iv.x; acc.y = beta * iv.y; acc.z = beta * iv.z; acc.w = beta * iv.w;
  }
  for (int j = jbeg; j < jend; ++j) {
    const int e = perm[j];
    const float a = alpha * vals[e];
    const float4 g = *(const float4*)(in + (size_t)col[e] * C + c4);
    acc.x += a * g.x; acc.y += a * g.y; acc.z += a * g.z; acc.w += a * g.w;
  }
  *(float4*)(out + (size_t)v * C + c4) = acc;
}

// ------------------------- group norm -------------------------
__global__ void gn_stats_k(const float* __restrict__ x, float* __restrict__ stats,
                           int nrows, int gshift) {
  __shared__ float sh[16];
  const int c = threadIdx.x;
  const int C = blockDim.x;
  if (c < 16) sh[c] = 0.0f;
  __syncthreads();
  float s = 0.0f, ss = 0.0f;
  for (int v = blockIdx.x; v < nrows; v += gridDim.x) {
    float t = x[(size_t)v * C + c];
    s += t;
    ss += t * t;
  }
  const int g = c >> gshift;
  atomicAdd(&sh[2 * g + 0], s);
  atomicAdd(&sh[2 * g + 1], ss);
  __syncthreads();
  if (c < 16) atomicAdd(&stats[c], sh[c]);
}

__global__ void gn_apply_silu_k(const float* __restrict__ x, float* __restrict__ y,
                                const float* __restrict__ stats,
                                const float* __restrict__ gamma,
                                const float* __restrict__ beta,
                                unsigned cmask, int gshift, float invN, size_t n) {
  size_t stride = (size_t)gridDim.x * blockDim.x;
  for (size_t i = (size_t)blockIdx.x * blockDim.x + threadIdx.x; i < n; i += stride) {
    unsigned c = (unsigned)i & cmask;
    int g = (int)(c >> gshift);
    float mu  = stats[2 * g + 0] * invN;
    float var = stats[2 * g + 1] * invN - mu * mu;
    float t = (x[i] - mu) * rsqrtf(var + EPSG) * gamma[c] + beta[c];
    y[i] = silu_f(t);
  }
}

// ------------------------- time-embedding MLP -------------------------
__global__ void time_mlp_k(const float* __restrict__ te, const float* __restrict__ wt,
                           const float* __restrict__ bt, float* __restrict__ tbias) {
  __shared__ float sh[256];
  const int t = threadIdx.x;
  sh[t] = silu_f(te[t]);
  __syncthreads();
  float s = bt[t];
  #pragma unroll 8
  for (int k = 0; k < 256; ++k) s += sh[k] * wt[k * COUT + t];
  tbias[t] = s;
}

// ------------------------- WMMA GEMM: acc[V,F] += T[V,C] * W[C,F] -------------------------
// grid = (V/128, F/64); 256 threads = 8 waves; each wave owns a 16x64 output strip
// (rows blockIdx.x*128 + wave*16, cols blockIdx.y*64): one A fragment feeds 4 WMMAs.
__global__ __launch_bounds__(256) void wmma_gemm_acc_k(
    const float* __restrict__ T, const float* __restrict__ W,
    float* __restrict__ acc, int C, int F) {
  __shared__ bf16_t Bs[256 * 64];  // up to C=256 rows x 64 cols, 32 KB
  const int tid  = threadIdx.x;
  const int col0 = blockIdx.y * 64;
  for (int i = tid; i < C * 32; i += 256) {
    const int c = i >> 5, f2 = (i & 31) << 1;
    const float2 wv = *(const float2*)(W + (size_t)c * F + col0 + f2);
    Bs[c * 64 + f2 + 0] = (bf16_t)wv.x;
    Bs[c * 64 + f2 + 1] = (bf16_t)wv.y;
  }
  __syncthreads();

  const int wave = tid >> 5;
  const int lane = tid & 31;
  const int row0 = blockIdx.x * 128 + wave * 16;
  const int m  = lane & 15;   // row (A) / col (C,D) within tile
  const int kh = lane >> 4;   // K-half selector for A
  v8f d0 = {}, d1 = {}, d2 = {}, d3 = {};
  const float* arow = T + (size_t)(row0 + m) * C;

  for (int kk = 0; kk < C; kk += 32) {
    v16bf a;
    // A 16x32 bf16: lane<16 -> M=lane, elems 0..7 = K 0..7, 8..15 = K 16..23
    //               lane>=16 -> M=lane-16, elems 0..7 = K 8..15, 8..15 = K 24..31
    const float* ap = arow + kk + kh * 8;
    #pragma unroll
    for (int j = 0; j < 8; ++j) {
      a[j]     = (bf16_t)ap[j];
      a[j + 8] = (bf16_t)ap[16 + j];
    }
    // B 32x16 bf16: N = lane%16; elems 0..7 = K 8*kh+0..7, 8..15 = K 16+8*kh+0..7 (per 16-col tile)
    const bf16_t* bp = Bs + (kk + kh * 8) * 64 + m;
    v16bf b0, b1, b2, b3;
    #pragma unroll
    for (int j = 0; j < 8; ++j) {
      b0[j] = bp[j * 64];
      b1[j] = bp[j * 64 + 16];
      b2[j] = bp[j * 64 + 32];
      b3[j] = bp[j * 64 + 48];
      b0[j + 8] = bp[(16 + j) * 64];
      b1[j + 8] = bp[(16 + j) * 64 + 16];
      b2[j + 8] = bp[(16 + j) * 64 + 32];
      b3[j + 8] = bp[(16 + j) * 64 + 48];
    }
    d0 = __builtin_amdgcn_wmma_f32_16x16x32_bf16(false, a, false, b0, (short)0, d0, false, false);
    d1 = __builtin_amdgcn_wmma_f32_16x16x32_bf16(false, a, false, b1, (short)0, d1, false, false);
    d2 = __builtin_amdgcn_wmma_f32_16x16x32_bf16(false, a, false, b2, (short)0, d2, false, false);
    d3 = __builtin_amdgcn_wmma_f32_16x16x32_bf16(false, a, false, b3, (short)0, d3, false, false);
  }

  // C/D 16x16 f32: lane<16 -> (M=r, N=lane); lane>=16 -> (M=r+8, N=lane-16)
  float* op = acc + (size_t)(row0 + (kh ? 8 : 0)) * F + col0 + m;
  #pragma unroll
  for (int r = 0; r < 8; ++r) {
    op[(size_t)r * F +  0] += d0[r];
    op[(size_t)r * F + 16] += d1[r];
    op[(size_t)r * F + 32] += d2[r];
    op[(size_t)r * F + 48] += d3[r];
  }
}

// ------------------------- host orchestration -------------------------
extern "C" void kernel_launch(void* const* d_in, const int* in_sizes, int n_in,
                              void* d_out, int out_size, void* d_ws, size_t ws_size,
                              hipStream_t stream) {
  (void)in_sizes; (void)n_in; (void)out_size; (void)ws_size;

  const float* x    = (const float*)d_in[0];
  const float* temb = (const float*)d_in[1];
  const int*   row  = (const int*)d_in[2];
  const int*   col  = (const int*)d_in[3];
  const float* vals = (const float*)d_in[4];
  const float* wres = (const float*)d_in[5];
  const float* w1   = (const float*)d_in[6];
  const float* w2   = (const float*)d_in[7];
  const float* g1   = (const float*)d_in[8];
  const float* b1   = (const float*)d_in[9];
  const float* g2   = (const float*)d_in[10];
  const float* b2   = (const float*)d_in[11];
  const float* wt   = (const float*)d_in[12];
  const float* bt   = (const float*)d_in[13];
  float* out = (float*)d_out;

  // ---- workspace carve-up (32-bit slots) ----
  float* ws = (float*)d_ws;
  float* stats1 = ws + 0;    // 16
  float* stats2 = ws + 16;   // 16
  float* tbias  = ws + 64;   // 256
  size_t P = 1024;
  float* norm_x = ws + P; P += (size_t)V * CIN;   // silu(gn1(x)); recycled as T2,T4,... of block1
  float* bufA   = ws + P; P += (size_t)V * CIN;   // ping-pong (C=128)
  float* bufB   = ws + P; P += (size_t)V * CIN;   // ping-pong (C=128, residual conv only)
  float* acc1   = ws + P; P += (size_t)V * COUT;  // h after block1 + time bias
  float* normH  = ws + P; P += (size_t)V * COUT;  // silu(gn2(h)); recycled as T2,T4,... of block2
  float* buf256 = ws + P; P += (size_t)V * COUT;  // ping-pong (C=256)
  int* counts   = (int*)(ws + P); P += V;         // also reused as scatter cursor
  int* rowptr   = (int*)(ws + P); P += V + 64;
  int* perm     = (int*)(ws + P); P += EDG;

  auto zero = [&](float* p, size_t n) {
    fill_zero_k<<<2048, 256, 0, stream>>>(p, n);
  };
  auto gemm = [&](const float* T, const float* Wk, float* acc, int C) {
    dim3 g(V / 128, COUT / 64);
    wmma_gemm_acc_k<<<g, 256, 0, stream>>>(T, Wk, acc, C, COUT);
  };
  auto spmv = [&](float* o, const float* in, const float* initsrc, float beta, float alpha,
                  int C) {
    const int cshift = (C == 128) ? 5 : 6;
    const unsigned blocks = (unsigned)(((size_t)V * (C >> 2) + 255) / 256);
    spmv_csr_k<<<blocks, 256, 0, stream>>>(o, in, initsrc, beta, alpha,
                                           rowptr, perm, col, vals, cshift);
  };
  // acc += sum_k T_k * w[k];  T_k = 2 L T_{k-1} - T_{k-2}, where the init term
  // (-T_{k-2}) is fused into the gather SpMV (beta = -1, alpha = 2).
  auto cheb = [&](const float* T0, float* buf1, float* buf2, bool t0_writable,
                  const float* w, float* acc, int C) {
    const size_t wstride = (size_t)C * COUT;
    gemm(T0, w, acc, C);                                 // k = 0
    spmv(buf1, T0, T0, 0.0f, 1.0f, C);                   // T1 = L T0
    gemm(buf1, w + wstride, acc, C);                     // k = 1
    float* cur = buf1;
    float* nxt = t0_writable ? (float*)T0 : buf2;
    for (int k = 2; k < KCH; ++k) {
      const float* initsrc = (k == 2) ? T0 : nxt;        // nxt holds T_{k-2} for k > 2
      spmv(nxt, cur, initsrc, -1.0f, 2.0f, C);           // T_k = 2 L T_{k-1} - T_{k-2}
      gemm(nxt, w + (size_t)k * wstride, acc, C);
      float* t = cur; cur = nxt; nxt = t;
    }
  };

  // ---- build CSR permutation of the edge list (cheap, rebuilt every call) ----
  fill_zero_k<<<256, 256, 0, stream>>>((float*)counts, V);
  count_rows_k<<<1728, 256, 0, stream>>>(row, counts, EDG);
  rowptr_scan_k<<<1, 1024, 0, stream>>>(counts, rowptr);
  hipMemcpyAsync(counts, rowptr, (size_t)V * sizeof(int),
                 hipMemcpyDeviceToDevice, stream);        // counts -> cursor
  scatter_perm_k<<<1728, 256, 0, stream>>>(row, counts, perm, EDG);

  // ---- pipeline ----
  zero(ws, 1024);  // stats1, stats2

  // gn1 + silu
  gn_stats_k<<<512, CIN, 0, stream>>>(x, stats1, V, 4);
  const float invN1 = 1.0f / ((float)V * (CIN / 8));
  gn_apply_silu_k<<<2048, 256, 0, stream>>>(x, norm_x, stats1, g1, b1,
                                            CIN - 1, 4, invN1, (size_t)V * CIN);

  // time embedding bias
  time_mlp_k<<<1, 256, 0, stream>>>(temb, wt, bt, tbias);

  // residual conv -> out (out doubles as final accumulator)
  zero(out, (size_t)V * COUT);
  cheb(x, bufA, bufB, /*t0_writable=*/false, wres, out, CIN);

  // block1 -> acc1, pre-seeded with broadcast time bias (h = conv + tbias)
  bcast_row_k<<<2048, 256, 0, stream>>>(acc1, tbias, COUT - 1, (size_t)V * COUT);
  cheb(norm_x, bufA, nullptr, /*t0_writable=*/true, w1, acc1, CIN);

  // gn2 + silu on h
  gn_stats_k<<<512, COUT, 0, stream>>>(acc1, stats2, V, 5);
  const float invN2 = 1.0f / ((float)V * (COUT / 8));
  gn_apply_silu_k<<<2048, 256, 0, stream>>>(acc1, normH, stats2, g2, b2,
                                            COUT - 1, 5, invN2, (size_t)V * COUT);

  // block2 -> out (already contains res, so final "+ res" is fused)
  cheb(normH, buf256, nullptr, /*t0_writable=*/true, w2, out, COUT);
}
